// Gird_52974126629188
// MI455X (gfx1250) — matrix operand order repaired and checked by reading, not compiled, not run on hardware
//
#include <hip/hip_runtime.h>

typedef __attribute__((ext_vector_type(16))) _Float16 v16h;
typedef __attribute__((ext_vector_type(8)))  float    v8f;

#define HH     180
#define WW     240
#define T_AGG  3
#define NB     4
#define M_EV   32
#define MLP_H  20
#define RNN_H  20
#define LRELU  0.1f
#define ROWS   (NB * T_AGG * 2 * HH * WW)   // 1,036,800

__device__ __forceinline__ void wave_lds_sync() {
    // make this wave's LDS writes visible to all its lanes before re-reading
    asm volatile("s_wait_dscnt 0x0" ::: "memory");
    __builtin_amdgcn_wave_barrier();
}

__device__ __forceinline__ float lrelu(float x) { return x > 0.f ? x : LRELU * x; }
__device__ __forceinline__ float sigm(float x)  { return 1.f / (1.f + __expf(-x)); }
__device__ __forceinline__ int   imin(int a, int b) { return a < b ? a : b; }

// ---------------------------------------------------------------- zero vox
__global__ void __launch_bounds__(256) zero_vox(float4* __restrict__ v, int n4) {
    int i = blockIdx.x * 256 + threadIdx.x;
    if (i < n4) v[i] = make_float4(0.f, 0.f, 0.f, 0.f);
}

// ------------------------------------------------- per-event MLP + scatter
// 1 wave = 16 events. Layer2 (20x20, K padded to 32) via v_wmma_f32_16x16x32_f16.
__global__ void __launch_bounds__(256) mlp_scatter(
    const float* __restrict__ ev,
    const float* __restrict__ W1, const float* __restrict__ b1,
    const float* __restrict__ W2, const float* __restrict__ b2,
    const float* __restrict__ W3, const float* __restrict__ b3,
    float* __restrict__ vox, int nev)
{
    __shared__ float h2buf[8][16][32];     // padded to 32 cols: unconditional stores
    const int lane = threadIdx.x & 31;
    const int wave = threadIdx.x >> 5;
    const int m    = lane & 15;            // event row within tile
    const int hg   = (lane < 16) ? 0 : 1;  // lane half-group
    const long e0  = ((long)blockIdx.x * 8 + wave) * 16;
    const long e   = e0 + m;
    const bool valid = (e < (long)nev);

    // prefetch next block's event tile (global_prefetch_b8; no counter)
    __builtin_prefetch(ev + (e0 + 128) * 7, 0, 0);

    float t = valid ? ev[e * 7 + 3] : 0.f;

    // layer 1 (1->20) computed per-lane directly into WMMA A layout (16x32 f16)
    v16h a;
#pragma unroll
    for (int hi = 0; hi < 16; ++hi) {
        int k  = (hi < 8 ? hi : hi + 8) + hg * 8;   // A layout K index
        int kc = imin(k, MLP_H - 1);                // clamp: branch-free load
        float h1 = lrelu(t * W1[kc] + b1[kc]);
        a[hi] = (_Float16)((k < MLP_H) ? h1 : 0.f);
    }

    // layer 2 (20->20): two N-tiles of 16
#pragma unroll
    for (int nt = 0; nt < 2; ++nt) {
        int n  = nt * 16 + m;
        int nc = imin(n, MLP_H - 1);
        v16h bmat;
#pragma unroll
        for (int hi = 0; hi < 16; ++hi) {
            int k  = (hi < 8 ? hi : hi + 8) + hg * 8;   // B layout K index
            int kc = imin(k, MLP_H - 1);
            float w = W2[nc * MLP_H + kc];
            bmat[hi] = (_Float16)((k < MLP_H && n < MLP_H) ? w : 0.f);
        }
        v8f c = {};
        c = __builtin_amdgcn_wmma_f32_16x16x32_f16(false, a, false, bmat,
                                                   (short)0, c, false, false);
        float bias = b2[nc];
#pragma unroll
        for (int i = 0; i < 8; ++i) {
            int mr = i + hg * 8;                    // C layout row
            h2buf[wave][mr][n] = lrelu(c[i] + bias);
        }
    }
    wave_lds_sync();

    // layer 3 (20->1) + scatter; lanes 0..15 own one event each
    if (lane < 16 && valid) {
        float acc = b3[0];
#pragma unroll
        for (int j = 0; j < MLP_H; ++j) acc += h2buf[wave][lane][j] * W3[j];
        const float* er = ev + e * 7;
        int x  = (int)er[0], y = (int)er[1], p = (int)er[2];
        int ip = (int)er[4], tm = (int)er[5], b = (int)er[6];
        long row = x + (long)WW * (y + (long)HH * (p + 2 * (b * T_AGG + tm)));
        vox[row * M_EV + (ip - 1)] = acc;          // indices unique by construction
    }
}

// ------------------------------------------------------------- LSTM rows
// 1 wave = 16 rows; h(16x20) @ W_hh.T(20x80) per step via 5 WMMA N-tiles.
// Runs only max(length) steps per tile (state frozen when t >= length[row],
// identical to the reference's masked 32-step scan).
__global__ void __launch_bounds__(128) lstm_rows(
    const float* __restrict__ vox,
    const float* __restrict__ rW1, const float* __restrict__ rb1,
    const float* __restrict__ W_ih, const float* __restrict__ W_hh,
    const float* __restrict__ b_ih, const float* __restrict__ b_hh,
    const float* __restrict__ rW2, const float* __restrict__ rb2,
    float* __restrict__ out)
{
    __shared__ float uu[80], vv[80];
    __shared__ float xbuf[4][16][M_EV];
    __shared__ float hbuf[4][16][RNN_H];
    __shared__ float cbuf[4][16][RNN_H];
    __shared__ float gbuf[4][16][4 * RNN_H];
    __shared__ float sumb[4][16];
    __shared__ int   lenb[4][16];

    const int tid = threadIdx.x;
    // fold rank-1 input path: gates += x*u + v  (u = W_ih@rW1, v = W_ih@rb1 + b_ih + b_hh)
    if (tid < 4 * RNN_H) {
        float u = 0.f, w = 0.f;
#pragma unroll
        for (int j = 0; j < RNN_H; ++j) {
            float wij = W_ih[tid * RNN_H + j];
            u += wij * rW1[j];
            w += wij * rb1[j];
        }
        uu[tid] = u;
        vv[tid] = w + b_ih[tid] + b_hh[tid];
    }
    __syncthreads();   // uniform: grid is exact, no wave exits before this

    const int lane = tid & 31;
    const int wave = tid >> 5;
    const int m    = lane & 15;
    const int hg   = (lane < 16) ? 0 : 1;
    const long r0  = ((long)blockIdx.x * 4 + wave) * 16;

    // prefetch next block's vox tiles
    __builtin_prefetch(vox + (r0 + 64 + m) * M_EV, 0, 0);

    // W_hh.T in WMMA B layout, 5 N-tiles of 16 gates each (K=20 padded to 32)
    v16h Bw[5];
#pragma unroll
    for (int nt = 0; nt < 5; ++nt) {
        int g = nt * 16 + m;
#pragma unroll
        for (int hi = 0; hi < 16; ++hi) {
            int k  = (hi < 8 ? hi : hi + 8) + hg * 8;
            int kc = imin(k, RNN_H - 1);
            float w = W_hh[g * RNN_H + kc];
            Bw[nt][hi] = (_Float16)((k < RNN_H) ? w : 0.f);
        }
    }

    // load vox tile (each lane: row m, 16-float half), length + sum
    {
        const float4* vp = (const float4*)(vox + (r0 + m) * M_EV + hg * 16);
        float psum = 0.f; int plen = 0;
#pragma unroll
        for (int q = 0; q < 4; ++q) {
            float4 d = vp[q];
            int kk = hg * 16 + q * 4;
            xbuf[wave][m][kk + 0] = d.x; xbuf[wave][m][kk + 1] = d.y;
            xbuf[wave][m][kk + 2] = d.z; xbuf[wave][m][kk + 3] = d.w;
            psum += d.x + d.y + d.z + d.w;
            plen += (d.x != 0.f) + (d.y != 0.f) + (d.z != 0.f) + (d.w != 0.f);
        }
        psum += __shfl_xor(psum, 16, 32);
        plen += __shfl_xor(plen, 16, 32);
        if (lane < 16) { sumb[wave][m] = psum; lenb[wave][m] = plen; }
    }
    // zero-init h, c (320 elements / 32 lanes)
#pragma unroll
    for (int q = 0; q < 10; ++q) {
        int e = lane + 32 * q;
        hbuf[wave][e / RNN_H][e % RNN_H] = 0.f;
        cbuf[wave][e / RNN_H][e % RNN_H] = 0.f;
    }
    wave_lds_sync();

    int maxlen = lenb[wave][m];
#pragma unroll
    for (int off = 8; off >= 1; off >>= 1)
        maxlen = max(maxlen, __shfl_xor(maxlen, off, 32));

    for (int t = 0; t < maxlen; ++t) {
        // A = h (f32 -> f16), 16x32 A layout; clamp+select: branch-free LDS reads
        v16h a;
#pragma unroll
        for (int hi = 0; hi < 16; ++hi) {
            int k  = (hi < 8 ? hi : hi + 8) + hg * 8;
            int kc = imin(k, RNN_H - 1);
            float hv = hbuf[wave][m][kc];
            a[hi] = (_Float16)((k < RNN_H) ? hv : 0.f);
        }
        // gates = h @ W_hh.T + x*u + v, written to LDS in [m][gate] order
#pragma unroll
        for (int nt = 0; nt < 5; ++nt) {
            v8f c = {};
            c = __builtin_amdgcn_wmma_f32_16x16x32_f16(false, a, false, Bw[nt],
                                                       (short)0, c, false, false);
            int g = nt * 16 + m;
            float ug = uu[g], vg = vv[g];
#pragma unroll
            for (int i = 0; i < 8; ++i) {
                int mr = i + hg * 8;
                gbuf[wave][mr][g] = c[i] + xbuf[wave][mr][t] * ug + vg;
            }
        }
        wave_lds_sync();
        // activations + masked state update (mask is required per-row semantics)
#pragma unroll
        for (int q = 0; q < 10; ++q) {
            int e  = lane + 32 * q;
            int mr = e / RNN_H, j = e % RNN_H;
            if (t < lenb[wave][mr]) {
                float gi = gbuf[wave][mr][j];
                float gf = gbuf[wave][mr][RNN_H + j];
                float gg = gbuf[wave][mr][2 * RNN_H + j];
                float go = gbuf[wave][mr][3 * RNN_H + j];
                float cn = sigm(gf) * cbuf[wave][mr][j] + sigm(gi) * tanhf(gg);
                cbuf[wave][mr][j] = cn;
                hbuf[wave][mr][j] = sigm(go) * tanhf(cn);
            }
        }
        wave_lds_sync();
    }

    // output head: out = use_rnn ? h @ rW2.T + rb2 : 0
    if (lane < 16) {
        float acc = rb2[0];
#pragma unroll
        for (int j = 0; j < RNN_H; ++j) acc += hbuf[wave][m][j] * rW2[j];
        out[r0 + m] = (sumb[wave][m] != 0.f) ? acc : 0.f;
    }
}

// ----------------------------------------------------------------- launch
extern "C" void kernel_launch(void* const* d_in, const int* in_sizes, int n_in,
                              void* d_out, int out_size, void* d_ws, size_t ws_size,
                              hipStream_t stream)
{
    const float* ev     = (const float*)d_in[0];
    const float* mlp_W1 = (const float*)d_in[1];
    const float* mlp_b1 = (const float*)d_in[2];
    const float* mlp_W2 = (const float*)d_in[3];
    const float* mlp_b2 = (const float*)d_in[4];
    const float* mlp_W3 = (const float*)d_in[5];
    const float* mlp_b3 = (const float*)d_in[6];
    const float* rW1    = (const float*)d_in[7];
    const float* rb1    = (const float*)d_in[8];
    const float* W_ih   = (const float*)d_in[9];
    const float* W_hh   = (const float*)d_in[10];
    const float* b_ih   = (const float*)d_in[11];
    const float* b_hh   = (const float*)d_in[12];
    const float* rW2    = (const float*)d_in[13];
    const float* rb2    = (const float*)d_in[14];

    float* vox = (float*)d_ws;                       // ROWS * 32 floats (~132 MB)
    const int nev = in_sizes[0] / 7;

    const int n4 = ROWS * M_EV / 4;
    zero_vox<<<(n4 + 255) / 256, 256, 0, stream>>>((float4*)vox, n4);

    const int nwaves = (nev + 15) / 16;
    const int nblk   = (nwaves + 7) / 8;
    mlp_scatter<<<nblk, 256, 0, stream>>>(ev, mlp_W1, mlp_b1, mlp_W2, mlp_b2,
                                          mlp_W3, mlp_b3, vox, nev);

    lstm_rows<<<ROWS / 64, 128, 0, stream>>>(vox, rW1, rb1, W_ih, W_hh,
                                             b_ih, b_hh, rW2, rb2, (float*)d_out);
}
